// Attention_89979564851805
// MI455X (gfx1250) — matrix operand order repaired and checked by reading, not compiled, not run on hardware
//
#include <hip/hip_runtime.h>

typedef __bf16 bf16;
typedef __attribute__((ext_vector_type(8)))  __bf16 v8bf;
typedef __attribute__((ext_vector_type(16))) __bf16 v16bf;
typedef __attribute__((ext_vector_type(8)))  float  v8f;

// Load a 16-element bf16 fragment half-pair from two 16B-aligned LDS chunks.
static __device__ __forceinline__ v16bf ld_frag16(const bf16* p0, const bf16* p1) {
    v8bf lo = *(const v8bf*)p0;
    v8bf hi = *(const v8bf*)p1;
    return __builtin_shufflevector(lo, hi, 0,1,2,3,4,5,6,7,8,9,10,11,12,13,14,15);
}

static __device__ __forceinline__ v8f wmma_bf16(v16bf a, v16bf b, v8f c) {
    return __builtin_amdgcn_wmma_f32_16x16x32_bf16(
        false, a, false, b, (short)0, c, false, false);
}

// CDNA5 async copy: 16 bytes global -> LDS, tracked by ASYNCcnt.
static __device__ __forceinline__ void async_cp16(unsigned lds_off, const bf16* g) {
    asm volatile("global_load_async_to_lds_b128 %0, %1, off"
                 :: "v"(lds_off), "v"(g) : "memory");
}
#define S_WAIT_ASYNC(n) asm volatile("s_wait_asynccnt " #n ::: "memory")

// ---------------------------------------------------------------------------
// f32 -> bf16 elementwise convert
// ---------------------------------------------------------------------------
__global__ __launch_bounds__(256) void cvt_f32_bf16(const float* __restrict__ in,
                                                    bf16* __restrict__ out, long n) {
    long i = (long)blockIdx.x * 256 + threadIdx.x;
    if (i < n) out[i] = (bf16)in[i];
}

// ---------------------------------------------------------------------------
// C[M,N] (f32) = A[M,K] (bf16, row-major) * B[N,K]^T (bf16, row-major rows=N)
// Block: 128x128 tile, 256 threads = 8 waves as 2(M)x4(N); wave tile 64x32.
// BK = 32; LDS double-buffered, filled by global_load_async_to_lds_b128.
// ---------------------------------------------------------------------------
#define GEMM_AST 40   // 32 + 8 pad, stride*2B = 80B (16B multiple)

__global__ __launch_bounds__(256) void gemm_bf16_xwT(const bf16* __restrict__ A,
                                                     const bf16* __restrict__ B,
                                                     float* __restrict__ C,
                                                     int M, int N, int K) {
    __shared__ bf16 Ash[2][128 * GEMM_AST];
    __shared__ bf16 Bsh[2][128 * GEMM_AST];

    const int tid   = threadIdx.x;
    const int lane  = tid & 31;
    const int wid   = tid >> 5;
    const int wm    = wid >> 2;        // 0..1 : 64-row slab
    const int wn    = wid & 3;         // 0..3 : 32-col slab
    const int lhalf = lane >> 4;
    const int lrow  = lane & 15;

    const int m0 = blockIdx.y * 128;
    const int n0 = blockIdx.x * 128;

    v8f acc[4][2] = {};

    // cooperative loader: thread t -> row t>>1, 16 cols at (t&1)*16
    const int ar = tid >> 1;
    const int ac = (tid & 1) * 16;

    unsigned la[2], lb[2];
    la[0] = (unsigned)(uintptr_t)&Ash[0][ar * GEMM_AST + ac];
    la[1] = (unsigned)(uintptr_t)&Ash[1][ar * GEMM_AST + ac];
    lb[0] = (unsigned)(uintptr_t)&Bsh[0][ar * GEMM_AST + ac];
    lb[1] = (unsigned)(uintptr_t)&Bsh[1][ar * GEMM_AST + ac];
    const bf16* ga = A + (size_t)(m0 + ar) * K + ac;
    const bf16* gb = B + (size_t)(n0 + ar) * K + ac;

    // preload K-tile 0 into buffer 0
    async_cp16(la[0], ga);      async_cp16(la[0] + 16, ga + 8);
    async_cp16(lb[0], gb);      async_cp16(lb[0] + 16, gb + 8);

    const int nk = K >> 5;
    for (int kt = 0; kt < nk; ++kt) {
        const int cur = kt & 1;
        if (kt + 1 < nk) {
            const int nxt = cur ^ 1;
            const bf16* ga2 = ga + (size_t)(kt + 1) * 32;
            const bf16* gb2 = gb + (size_t)(kt + 1) * 32;
            async_cp16(la[nxt], ga2);  async_cp16(la[nxt] + 16, ga2 + 8);
            async_cp16(lb[nxt], gb2);  async_cp16(lb[nxt] + 16, gb2 + 8);
            S_WAIT_ASYNC(0x4);      // in-order completion: current tile landed
        } else {
            S_WAIT_ASYNC(0x0);
        }
        __syncthreads();            // all waves' current tile visible

        const bf16* Acur = Ash[cur];
        const bf16* Bcur = Bsh[cur];

        // B fragments: lane = N col, k = lhalf*16 + e
        v16bf bfrag[2];
        #pragma unroll
        for (int nt = 0; nt < 2; ++nt) {
            const bf16* p = &Bcur[(wn * 32 + nt * 16 + lrow) * GEMM_AST + lhalf * 16];
            bfrag[nt] = ld_frag16(p, p + 8);
        }
        #pragma unroll
        for (int mt = 0; mt < 4; ++mt) {
            // A fragment: lane = M row, k = lhalf*16 + e
            const bf16* p = &Acur[(wm * 64 + mt * 16 + lrow) * GEMM_AST + lhalf * 16];
            v16bf afrag = ld_frag16(p, p + 8);
            #pragma unroll
            for (int nt = 0; nt < 2; ++nt)
                acc[mt][nt] = wmma_bf16(afrag, bfrag[nt], acc[mt][nt]);
        }
        __syncthreads();            // all waves done reading before overwrite
    }

    // C layout: element g -> row g + 8*lhalf, col = lrow
    #pragma unroll
    for (int mt = 0; mt < 4; ++mt)
        #pragma unroll
        for (int nt = 0; nt < 2; ++nt) {
            const int col = n0 + wn * 32 + nt * 16 + lrow;
            #pragma unroll
            for (int g = 0; g < 8; ++g) {
                const int row = m0 + wm * 64 + mt * 16 + 8 * lhalf + g;
                C[(size_t)row * N + col] = acc[mt][nt][g];
            }
        }
}

// ---------------------------------------------------------------------------
// qkv f32 [B,S,3,nh,hd] -> Q/K/V bf16 [B*nh, S, hd]
// ---------------------------------------------------------------------------
__global__ __launch_bounds__(256) void split_qkv(const float* __restrict__ qkv,
                                                 bf16* __restrict__ Qo,
                                                 bf16* __restrict__ Ko,
                                                 bf16* __restrict__ Vo,
                                                 int B, int S, int nh, int hd) {
    long total = (long)B * nh * S * hd;
    long i = (long)blockIdx.x * 256 + threadIdx.x;
    if (i >= total) return;
    int d = (int)(i % hd);  long r = i / hd;
    int s = (int)(r % S);   r /= S;
    int h = (int)(r % nh);  int b = (int)(r / nh);
    const int HH = 3 * nh * hd;
    size_t src = (size_t)(b * S + s) * HH + h * hd + d;
    Qo[i] = (bf16)qkv[src];
    Ko[i] = (bf16)qkv[src + nh * hd];
    Vo[i] = (bf16)qkv[src + 2 * nh * hd];
}

// ---------------------------------------------------------------------------
// Causal flash attention, head_dim = 128.
// Grid: (S/128, B*nh). Block: 256 threads = 8 waves, wave = 16 query rows.
// Key blocks of 64; QK^T and P·V both via v_wmma_f32_16x16x32_bf16.
// K tile staged with async copies; V transposed manually into LDS.
// ---------------------------------------------------------------------------
#define KST 136   // 128 + 8 pad  (272B stride, 16B multiple)
#define VST  72   // 64 + 8 pad   (144B stride, 16B multiple)

__global__ __launch_bounds__(256) void flash_attn_hd128(const bf16* __restrict__ Q,
                                                        const bf16* __restrict__ Kk,
                                                        const bf16* __restrict__ V,
                                                        bf16* __restrict__ O,
                                                        int S, int nh) {
    __shared__ bf16 Ksh[64 * KST];        // [key][d]
    __shared__ bf16 Vt[128 * VST];        // [d][key]  (transposed)
    __shared__ bf16 Psh[8 * 16 * VST];    // per-wave [qrow][key]

    const int HD    = 128;
    const int bh    = blockIdx.y;
    const int qbase = blockIdx.x * 128;
    const int tid   = threadIdx.x;
    const int lane  = tid & 31;
    const int wid   = tid >> 5;
    const int lhalf = lane >> 4;
    const int lrow  = lane & 15;

    const size_t base = (size_t)bh * S * HD;

    // Q fragments (kept in registers): wave rows qbase + wid*16 + lrow
    v16bf qfrag[4];
    {
        const bf16* qp = Q + base + (size_t)(qbase + wid * 16 + lrow) * HD;
        #pragma unroll
        for (int dc = 0; dc < 4; ++dc) {
            const bf16* p = qp + dc * 32 + lhalf * 16;
            qfrag[dc] = ld_frag16(p, p + 8);
        }
    }

    v8f   acc_o[8] = {};
    float mrow[8], lsum[8];
    #pragma unroll
    for (int g = 0; g < 8; ++g) { mrow[g] = -1e30f; lsum[g] = 0.0f; }

    const float scale = 0.08838834764831845f;   // 1/sqrt(128)
    const int   qrow0 = qbase + wid * 16 + 8 * lhalf;
    const int   nkb   = blockIdx.x * 2 + 2;     // (qbase+128)/64 key blocks

    // loader assignment (constant per thread)
    const int lr  = tid >> 2;            // 0..63 (key row)
    const int lc0 = (tid & 3) * 32;      // d chunk
    const unsigned lk = (unsigned)(uintptr_t)&Ksh[lr * KST + lc0];

    for (int jb = 0; jb < nkb; ++jb) {
        const int kbase = jb * 64;
        // K tile: async global -> LDS (4 x 16B per thread)
        const bf16* kp = Kk + base + (size_t)(kbase + lr) * HD + lc0;
        async_cp16(lk,      kp);      async_cp16(lk + 16, kp + 8);
        async_cp16(lk + 32, kp + 16); async_cp16(lk + 48, kp + 24);
        // V tile: load + transpose into LDS
        const bf16* vp = V + base + (size_t)(kbase + lr) * HD + lc0;
        #pragma unroll
        for (int d = 0; d < 32; ++d)
            Vt[(lc0 + d) * VST + lr] = vp[d];
        S_WAIT_ASYNC(0x0);
        __syncthreads();

        // S = Q K^T : 4 column tiles of 16 keys, K-loop over 4 d-chunks
        v8f accs[4] = {};
        #pragma unroll
        for (int ct = 0; ct < 4; ++ct)
            #pragma unroll
            for (int dc = 0; dc < 4; ++dc) {
                const bf16* p = &Ksh[(ct * 16 + lrow) * KST + dc * 32 + lhalf * 16];
                accs[ct] = wmma_bf16(qfrag[dc], ld_frag16(p, p + 8), accs[ct]);
            }

        // scale + causal mask + row max
        float rmax[8];
        #pragma unroll
        for (int g = 0; g < 8; ++g) rmax[g] = -1e30f;
        #pragma unroll
        for (int ct = 0; ct < 4; ++ct) {
            const int key = kbase + ct * 16 + lrow;
            #pragma unroll
            for (int g = 0; g < 8; ++g) {
                float s = accs[ct][g] * scale;
                if (key > qrow0 + g) s = -1e30f;
                accs[ct][g] = s;
                rmax[g] = fmaxf(rmax[g], s);
            }
        }
        #pragma unroll
        for (int m = 1; m <= 8; m <<= 1)
            #pragma unroll
            for (int g = 0; g < 8; ++g)
                rmax[g] = fmaxf(rmax[g], __shfl_xor(rmax[g], m, 32));

        // online softmax update
        float alpha[8], psum[8];
        #pragma unroll
        for (int g = 0; g < 8; ++g) {
            const float mnew = fmaxf(mrow[g], rmax[g]);
            alpha[g] = __expf(mrow[g] - mnew);
            mrow[g]  = mnew;
            psum[g]  = 0.0f;
        }
        // P = exp(s - m): stash to per-wave LDS for C->A fragment re-layout
        bf16* pw = &Psh[wid * 16 * VST];
        #pragma unroll
        for (int ct = 0; ct < 4; ++ct)
            #pragma unroll
            for (int g = 0; g < 8; ++g) {
                const float p = __expf(accs[ct][g] - mrow[g]);
                psum[g] += p;
                pw[(8 * lhalf + g) * VST + ct * 16 + lrow] = (bf16)p;
            }
        #pragma unroll
        for (int m = 1; m <= 8; m <<= 1)
            #pragma unroll
            for (int g = 0; g < 8; ++g)
                psum[g] += __shfl_xor(psum[g], m, 32);
        #pragma unroll
        for (int g = 0; g < 8; ++g) lsum[g] = lsum[g] * alpha[g] + psum[g];
        #pragma unroll
        for (int nt = 0; nt < 8; ++nt)
            #pragma unroll
            for (int g = 0; g < 8; ++g)
                acc_o[nt][g] *= alpha[g];

        // O += P V : 2 key chunks of 32, 8 d-tiles of 16
        #pragma unroll
        for (int kc = 0; kc < 2; ++kc) {
            const bf16* pp = &pw[lrow * VST + kc * 32 + lhalf * 16];
            v16bf pfrag = ld_frag16(pp, pp + 8);
            #pragma unroll
            for (int nt = 0; nt < 8; ++nt) {
                const bf16* vv = &Vt[(nt * 16 + lrow) * VST + kc * 32 + lhalf * 16];
                acc_o[nt] = wmma_bf16(pfrag, ld_frag16(vv, vv + 8), acc_o[nt]);
            }
        }
        __syncthreads();
    }

    // epilogue: O / l, write bf16 [B,S,nh*hd]
    const int b = bh / nh, h = bh % nh;
    #pragma unroll
    for (int nt = 0; nt < 8; ++nt) {
        const int d = nt * 16 + lrow;
        #pragma unroll
        for (int g = 0; g < 8; ++g) {
            const int srow = qrow0 + g;
            O[((size_t)(b * S + srow)) * (nh * HD) + h * HD + d] =
                (bf16)(acc_o[nt][g] / lsum[g]);
        }
    }
}

// ---------------------------------------------------------------------------
extern "C" void kernel_launch(void* const* d_in, const int* in_sizes, int n_in,
                              void* d_out, int out_size, void* d_ws, size_t ws_size,
                              hipStream_t stream) {
    const float* hs   = (const float*)d_in[0];   // [2,2048,2048]
    const float* wqkv = (const float*)d_in[1];   // [6144,2048]
    const float* wo   = (const float*)d_in[2];   // [2048,2048]
    float* out = (float*)d_out;                  // [2,2048,2048] f32

    const int B = 2, S = 2048, H = 2048, NH = 16, HD = 128;
    const int M  = B * S;        // 4096
    const int N1 = 3 * H;        // 6144

    char* ws = (char*)d_ws;
    bf16*  x_bf    = (bf16*)ws;  ws += (size_t)M * H * 2;
    bf16*  wqkv_bf = (bf16*)ws;  ws += (size_t)N1 * H * 2;
    bf16*  wo_bf   = (bf16*)ws;  ws += (size_t)H * H * 2;
    float* qkv_f   = (float*)ws; ws += (size_t)M * N1 * 4;
    bf16*  Qb      = (bf16*)ws;  ws += (size_t)B * NH * S * HD * 2;
    bf16*  Kb      = (bf16*)ws;  ws += (size_t)B * NH * S * HD * 2;
    bf16*  Vb      = (bf16*)ws;  ws += (size_t)B * NH * S * HD * 2;
    bf16*  attn_bf = (bf16*)ws;  ws += (size_t)M * H * 2;

    const long nx = (long)M * H, nw1 = (long)N1 * H, nw2 = (long)H * H;
    cvt_f32_bf16<<<(unsigned)((nx  + 255) / 256), 256, 0, stream>>>(hs,   x_bf,    nx);
    cvt_f32_bf16<<<(unsigned)((nw1 + 255) / 256), 256, 0, stream>>>(wqkv, wqkv_bf, nw1);
    cvt_f32_bf16<<<(unsigned)((nw2 + 255) / 256), 256, 0, stream>>>(wo,   wo_bf,   nw2);

    gemm_bf16_xwT<<<dim3(N1 / 128, M / 128), 256, 0, stream>>>(x_bf, wqkv_bf, qkv_f, M, N1, H);

    const long nq = (long)B * NH * S * HD;
    split_qkv<<<(unsigned)((nq + 255) / 256), 256, 0, stream>>>(qkv_f, Qb, Kb, Vb, B, S, NH, HD);

    flash_attn_hd128<<<dim3(S / 128, B * NH), 256, 0, stream>>>(Qb, Kb, Vb, attn_bf, S, NH);

    gemm_bf16_xwT<<<dim3(H / 128, M / 128), 256, 0, stream>>>(attn_bf, wo_bf, out, M, H, H);
}
